// AdaptiveSoftmaxV1_7121055776931
// MI455X (gfx1250) — matrix operand order repaired and checked by reading, not compiled, not run on hardware
//
#include <hip/hip_runtime.h>

// ---------------------------------------------------------------------------
// CDNA5 (gfx1250) adaptive-softmax CE, bf16 WMMA path.
// Head GEMM-CE dominates (168 of 222 GFLOP): stream packed bf16 B tiles
// (L2-resident) through v_wmma_f32_16x16x32_bf16 with per-lane online LSE.
// Two column tiles per wave iteration reuse one LDS A-fragment (2 ds_load_b128
// + 4 global_load_b128 per 2 WMMAs).
// ---------------------------------------------------------------------------

typedef __attribute__((ext_vector_type(16))) __bf16 v16bf;
typedef __attribute__((ext_vector_type(8)))  __bf16 v8bf;
typedef __attribute__((ext_vector_type(8)))  float  v8f;

#define ASTRIDE 1032   // (1024 + 8) bf16 -> 516 dwords row stride: LDS conflict-free

__device__ __forceinline__ unsigned short f2bf(float f) {
  unsigned int u = __float_as_uint(f);
  u += 0x7FFFu + ((u >> 16) & 1u);               // round-to-nearest-even
  return (unsigned short)(u >> 16);
}

// ---------------------------------------------------------------------------
// 1) f32 -> bf16 elementwise (activations, row-major)
// ---------------------------------------------------------------------------
__global__ void cvt_kernel(const float* __restrict__ X,
                           unsigned short* __restrict__ Y, int n) {
  int i = (blockIdx.x * blockDim.x + threadIdx.x) * 4;
  if (i + 3 < n) {
    float4 v = *(const float4*)&X[i];
    ushort4 o;
    o.x = f2bf(v.x); o.y = f2bf(v.y); o.z = f2bf(v.z); o.w = f2bf(v.w);
    *(ushort4*)&Y[i] = o;
  } else {
    for (; i < n; ++i) Y[i] = f2bf(X[i]);
  }
}

// ---------------------------------------------------------------------------
// 2) Pack W[K][V] (f32, row-major) into bf16 WMMA-B fragment order:
//    PB[((vt*nsteps + ks)*512) + lane*16 + i] = W[ks*32 + (lane&16?8:0) + (i<8?i:i+8)][vt*16 + (lane&15)]
//    Lane's 16 values contiguous -> CE-loop B fetch is a coalesced 1KB/wave burst.
// ---------------------------------------------------------------------------
__global__ void packb_kernel(const float* __restrict__ W,
                             unsigned short* __restrict__ PB,
                             int K, int V, int Vpad) {
  int t = blockIdx.x * blockDim.x + threadIdx.x;
  int L = t & 31;
  int q = t >> 5;
  int nsteps = K >> 5;
  int total = (Vpad >> 4) * nsteps;
  if (q >= total) return;
  int ks = q % nsteps, vt = q / nsteps;
  int v  = (vt << 4) + (L & 15);
  int kb = (ks << 5) + ((L & 16) ? 8 : 0);
  unsigned short tmp[16];
#pragma unroll
  for (int i = 0; i < 16; ++i) {
    float val = (v < V) ? W[(size_t)(kb + (i < 8 ? i : i + 8)) * V + v] : 0.0f;
    tmp[i] = f2bf(val);
  }
  uint4* dst = (uint4*)(PB + (size_t)q * 512 + L * 16);
  dst[0] = *(const uint4*)&tmp[0];
  dst[1] = *(const uint4*)&tmp[8];
}

// ---------------------------------------------------------------------------
// Shared A-tile loader: 16 rows x K bf16, row-major in LDS with pad.
// ---------------------------------------------------------------------------
__device__ __forceinline__ void load_A_tile(unsigned short* As,
                                            const unsigned short* __restrict__ A,
                                            int row0, int K, int tid) {
  int chunks = 2 * K;              // 16*K/8 chunks of 8 bf16 (16B)
  int cpr    = K >> 3;
  for (int c = tid; c < chunks; c += 256) {
    int r = c / cpr, o = (c % cpr) << 3;
    *(uint4*)&As[r * ASTRIDE + o] = *(const uint4*)&A[(size_t)(row0 + r) * K + o];
  }
}

// A fragment from LDS (16-bit A layout: lanes 0-15 row M=lane, K {0..7,16..23};
// lanes 16-31 same rows, K {8..15,24..31})
__device__ __forceinline__ v16bf load_A_frag(const unsigned short* As,
                                             int arow, int kbase) {
  v8bf a0 = *(const v8bf*)&As[arow * ASTRIDE + kbase];
  v8bf a1 = *(const v8bf*)&As[arow * ASTRIDE + kbase + 16];
  return __builtin_shufflevector(a0, a1, 0,1,2,3,4,5,6,7,8,9,10,11,12,13,14,15);
}

// ---------------------------------------------------------------------------
// 3) Projection GEMM: Out[N][P] (bf16) = A[N][K] (bf16) @ (packed B)
//    block = 16 rows; 8 waves stride over pairs of column tiles (P % 32 == 0).
// ---------------------------------------------------------------------------
__global__ __launch_bounds__(256) void proj_kernel(
    const unsigned short* __restrict__ A, const unsigned short* __restrict__ PB,
    unsigned short* __restrict__ Out, int K, int P) {
  __shared__ unsigned short As[16 * ASTRIDE];
  const int tid  = threadIdx.x;
  const int lane = tid & 31;
  const int wv   = tid >> 5;
  const int row0 = blockIdx.x * 16;

  load_A_tile(As, A, row0, K, tid);
  __syncthreads();

  const int nsteps = K >> 5;
  const int arow   = lane & 15;
  const int akoff  = (lane & 16) ? 8 : 0;
  const int rbase  = (lane & 16) ? 8 : 0;

  for (int vp = wv; vp < (P >> 5); vp += 8) {
    const int vt0 = vp * 2;
    v8f acc0 = {}, acc1 = {};
    const unsigned short* bp0 = PB + (size_t)vt0 * nsteps * 512 + lane * 16;
    const unsigned short* bp1 = bp0 + (size_t)nsteps * 512;
#pragma unroll 2
    for (int ks = 0; ks < nsteps; ++ks) {
      v16bf a  = load_A_frag(As, arow, (ks << 5) + akoff);
      v16bf b0 = *(const v16bf*)(bp0 + (size_t)ks * 512);
      v16bf b1 = *(const v16bf*)(bp1 + (size_t)ks * 512);
      acc0 = __builtin_amdgcn_wmma_f32_16x16x32_bf16(false, a, false, b0,
                                                     (short)0, acc0, false, false);
      acc1 = __builtin_amdgcn_wmma_f32_16x16x32_bf16(false, a, false, b1,
                                                     (short)0, acc1, false, false);
    }
    int col0 = (vt0 << 4) + (lane & 15);
#pragma unroll
    for (int j = 0; j < 8; ++j) {
      Out[(size_t)(row0 + rbase + j) * P + col0]      = f2bf(acc0[j]);
      Out[(size_t)(row0 + rbase + j) * P + col0 + 16] = f2bf(acc1[j]);
    }
  }
}

// ---------------------------------------------------------------------------
// Per-tile online-LSE fold: m/s/lb are per-lane accumulators for 8 rows.
// ---------------------------------------------------------------------------
__device__ __forceinline__ void online_fold(const v8f& acc, int col, int V,
                                            const int* rlab,
                                            float* m, float* s, float* lb) {
  const bool  valid = col < V;
  const float wgt   = valid ? 1.0f : 0.0f;   // pad columns contribute 0 to sumexp
#pragma unroll
  for (int j = 0; j < 8; ++j) {
    float v    = acc[j];
    float cand = valid ? v : -1e30f;
    float nm   = fmaxf(m[j], cand);
    s[j] = s[j] * __expf(m[j] - nm) + wgt * __expf(v - nm);
    m[j] = nm;
    if (valid && col == rlab[j]) lb[j] = v;
  }
}

// ---------------------------------------------------------------------------
// 4) Streaming CE: per-row online logsumexp over V logits + label-logit gather.
//    mode: 0 = tail0 (cutoff 20000), 1 = tail1 (cutoff 40000), 2 = head.
//    Vpad % 32 == 0 (pair loop).
// ---------------------------------------------------------------------------
__global__ __launch_bounds__(256) void ce_kernel(
    const unsigned short* __restrict__ A, const unsigned short* __restrict__ PB,
    const int* __restrict__ labels, float* __restrict__ out,
    int K, int V, int Vpad, int mode) {
  __shared__ unsigned short As[16 * ASTRIDE];
  __shared__ float red_m[8][16], red_s[8][16], red_l[8][16];
  __shared__ int tgt_s[16], msk_s[16];

  const int tid  = threadIdx.x;
  const int lane = tid & 31;
  const int wv   = tid >> 5;
  const int row0 = blockIdx.x * 16;

  if (tid < 16) {
    int lab = labels[row0 + tid];
    int tgt, msk;
    if (mode == 2) {             // head: remap tail tokens to cluster ids
      tgt = lab < 20000 ? lab : (lab < 40000 ? 20000 : 20001);
      msk = 1;
    } else {
      int cut = (mode == 0) ? 20000 : 40000;
      int t = lab - cut;
      tgt = t < 0 ? 0 : (t > V - 1 ? V - 1 : t);
      msk = (lab >= cut && lab < cut + 20000) ? 1 : 0;
    }
    tgt_s[tid] = tgt; msk_s[tid] = msk;
  }
  load_A_tile(As, A, row0, K, tid);
  __syncthreads();

  const int rbase = (lane & 16) ? 8 : 0;
  int rlab[8];
#pragma unroll
  for (int j = 0; j < 8; ++j) rlab[j] = tgt_s[rbase + j];

  float m[8], s[8], lb[8];
#pragma unroll
  for (int j = 0; j < 8; ++j) { m[j] = -1e30f; s[j] = 0.f; lb[j] = 0.f; }

  const int nsteps = K >> 5;
  const int arow   = lane & 15;
  const int akoff  = (lane & 16) ? 8 : 0;
  const int npairs = Vpad >> 5;

  for (int vp = wv; vp < npairs; vp += 8) {
    const int vt0  = vp * 2;
    const int col0 = (vt0 << 4) + (lane & 15);
    const unsigned short* bp0 = PB + (size_t)vt0 * nsteps * 512 + lane * 16;
    const unsigned short* bp1 = bp0 + (size_t)nsteps * 512;
    __builtin_prefetch(bp0 + (size_t)16 * nsteps * 512, 0, 0);  // next pair for this wave
    v8f acc0 = {}, acc1 = {};
#pragma unroll 2
    for (int ks = 0; ks < nsteps; ++ks) {
      v16bf a  = load_A_frag(As, arow, (ks << 5) + akoff);   // shared by both tiles
      v16bf b0 = *(const v16bf*)(bp0 + (size_t)ks * 512);
      v16bf b1 = *(const v16bf*)(bp1 + (size_t)ks * 512);
      acc0 = __builtin_amdgcn_wmma_f32_16x16x32_bf16(false, a, false, b0,
                                                     (short)0, acc0, false, false);
      acc1 = __builtin_amdgcn_wmma_f32_16x16x32_bf16(false, a, false, b1,
                                                     (short)0, acc1, false, false);
    }
    online_fold(acc0, col0,      V, rlab, m, s, lb);
    online_fold(acc1, col0 + 16, V, rlab, m, s, lb);
  }

  // butterfly LSE-merge within each 16-lane group (a C row spans 16 lanes)
#pragma unroll
  for (int j = 0; j < 8; ++j) {
#pragma unroll
    for (int d = 1; d < 16; d <<= 1) {
      float om = __shfl_xor(m[j], d);
      float os = __shfl_xor(s[j], d);
      float ol = __shfl_xor(lb[j], d);
      float nm = fmaxf(m[j], om);
      s[j] = s[j] * __expf(m[j] - nm) + os * __expf(om - nm);
      m[j] = nm;
      lb[j] += ol;
    }
  }
  if ((lane & 15) == 0) {
#pragma unroll
    for (int j = 0; j < 8; ++j) {
      int r = rbase + j;
      red_m[wv][r] = m[j]; red_s[wv][r] = s[j]; red_l[wv][r] = lb[j];
    }
  }
  __syncthreads();

  if (tid < 16) {
    float M = -1e30f, S = 0.f, L = 0.f;
#pragma unroll
    for (int w2 = 0; w2 < 8; ++w2) {
      float om = red_m[w2][tid], os = red_s[w2][tid];
      float nm = fmaxf(M, om);
      S = S * __expf(M - nm) + os * __expf(om - nm);
      M = nm;
      L += red_l[w2][tid];
    }
    float loss = (M + __logf(S)) - L;
    out[row0 + tid] = msk_s[tid] ? loss : 0.0f;
  }
}

// ---------------------------------------------------------------------------
// Host launcher
// ---------------------------------------------------------------------------
static inline size_t align256(size_t x) { return (x + 255) & ~(size_t)255; }

extern "C" void kernel_launch(void* const* d_in, const int* in_sizes, int n_in,
                              void* d_out, int out_size, void* d_ws, size_t ws_size,
                              hipStream_t stream) {
  const float* X      = (const float*)d_in[0];   // [N, H]
  const int*   labels = (const int*)d_in[1];     // [N]
  const float* Wh     = (const float*)d_in[2];   // [H, 20002]
  const float* P0     = (const float*)d_in[3];   // [H, 256]
  const float* W0     = (const float*)d_in[4];   // [256, 20000]
  const float* P1     = (const float*)d_in[5];   // [H, 64]
  const float* W1     = (const float*)d_in[6];   // [64, 20000]
  float* out = (float*)d_out;

  const int N  = in_sizes[1];          // 4096
  const int H  = in_sizes[0] / N;      // 1024
  const int VH = 20002, VHp = 20032;   // head vocab (+2 cluster logits), padded to x32
  const int V0 = 20000, K0 = 256;     // 20000 % 32 == 0
  const int V1 = 20000, K1 = 64;

  char* ws = (char*)d_ws;
  size_t off = 0;
  unsigned short* Xb  = (unsigned short*)(ws + off); off += align256((size_t)N * H * 2);
  unsigned short* Whp = (unsigned short*)(ws + off); off += align256((size_t)VHp * H * 2);
  unsigned short* W0p = (unsigned short*)(ws + off); off += align256((size_t)V0 * K0 * 2);
  unsigned short* W1p = (unsigned short*)(ws + off); off += align256((size_t)V1 * K1 * 2);
  unsigned short* P0p = (unsigned short*)(ws + off); off += align256((size_t)K0 * H * 2);
  unsigned short* P1p = (unsigned short*)(ws + off); off += align256((size_t)K1 * H * 2);
  unsigned short* Xp0 = (unsigned short*)(ws + off); off += align256((size_t)N * K0 * 2);
  unsigned short* Xp1 = (unsigned short*)(ws + off); off += align256((size_t)N * K1 * 2);

  // --- 1) convert activations to bf16 ---
  {
    int n = N * H;
    int blocks = (n / 4 + 255) / 256;
    cvt_kernel<<<blocks, 256, 0, stream>>>(X, Xb, n);
  }
  // --- 2) pack all weight matrices into WMMA-B fragment order (bf16) ---
  {
    int q, blocks;
    q = (VHp / 16) * (H / 32);  blocks = (q * 32 + 255) / 256;
    packb_kernel<<<blocks, 256, 0, stream>>>(Wh, Whp, H, VH, VHp);
    q = (V0 / 16) * (K0 / 32);  blocks = (q * 32 + 255) / 256;
    packb_kernel<<<blocks, 256, 0, stream>>>(W0, W0p, K0, V0, V0);
    q = (V1 / 16) * (K1 / 32);  blocks = (q * 32 + 255) / 256;
    packb_kernel<<<blocks, 256, 0, stream>>>(W1, W1p, K1, V1, V1);
    q = (K0 / 16) * (H / 32);   blocks = (q * 32 + 255) / 256;
    packb_kernel<<<blocks, 256, 0, stream>>>(P0, P0p, H, K0, K0);
    q = (K1 / 16) * (H / 32);   blocks = (q * 32 + 255) / 256;
    packb_kernel<<<blocks, 256, 0, stream>>>(P1, P1p, H, K1, K1);
  }
  // --- 3) tail projections: Xp = X @ P (bf16 WMMA, f32 accum) ---
  proj_kernel<<<N / 16, 256, 0, stream>>>(Xb, P0p, Xp0, H, K0);
  proj_kernel<<<N / 16, 256, 0, stream>>>(Xb, P1p, Xp1, H, K1);

  // --- 4) streaming CE; output layout = [tail0 | tail1 | head] ---
  ce_kernel<<<N / 16, 256, 0, stream>>>(Xp0, W0p, labels, out,         K0, V0, V0,  0);
  ce_kernel<<<N / 16, 256, 0, stream>>>(Xp1, W1p, labels, out + N,     K1, V1, V1,  1);
  ce_kernel<<<N / 16, 256, 0, stream>>>(Xb,  Whp, labels, out + 2 * N, H,  VH, VHp, 2);
}
